// Block_12738873000104
// MI455X (gfx1250) — matrix operand order repaired and checked by reading, not compiled, not run on hardware
//
#include <hip/hip_runtime.h>
#include <hip/hip_bf16.h>

typedef __attribute__((ext_vector_type(16))) _Float16 v16h;
typedef __attribute__((ext_vector_type(8)))  float    v8f;
typedef __attribute__((ext_vector_type(4)))  int      v4i;

#define WMMA_F16(a, b, c) \
  __builtin_amdgcn_wmma_f32_16x16x32_f16(false, (a), false, (b), (short)0, (c), false, false)

union F16x16 { v16h v; float4 f4[2]; };

// ---------------------------------------------------------------------------
// Async global->LDS staging (CDNA5 GLOBAL_LOAD_ASYNC_TO_LDS_B128, ASYNCcnt),
// with a synchronous fallback if this toolchain doesn't expose the builtins.
// Probe-confirmed param0 type: 'int4 __device__ *' (addrspace(1) v4i*).
// ---------------------------------------------------------------------------
#ifndef __has_builtin
#define __has_builtin(x) 0
#endif
#if __has_builtin(__builtin_amdgcn_global_load_async_to_lds_b128) && \
    __has_builtin(__builtin_amdgcn_s_wait_asynccnt)
#define USE_ASYNC_LDS 1
typedef __attribute__((address_space(1))) v4i glb_v4i;
typedef __attribute__((address_space(3))) v4i lds_v4i;
#else
#define USE_ASYNC_LDS 0
#endif

// copy 8 f16 (16B) global -> LDS
__device__ __forceinline__ void cp16(_Float16* s, const _Float16* g) {
#if USE_ASYNC_LDS
  __builtin_amdgcn_global_load_async_to_lds_b128((glb_v4i*)g, (lds_v4i*)s, 0, 0);
#else
  *(float4*)s = *(const float4*)g;
#endif
}

template <int N>
__device__ __forceinline__ void wait_async() {
#if USE_ASYNC_LDS
  __builtin_amdgcn_s_wait_asynccnt(N);
#endif
}

// ---------------------------------------------------------------------------
// WMMA fragment loaders (ISA §7.12.2 layouts, wave32)
// A (16x32 f16): lane r=lane&15 is row M=r; half h=lane>>4:
//   elements 0..7  hold K = 8h .. 8h+7        (16 contiguous bytes)
//   elements 8..15 hold K = 16+8h .. 16+8h+7  (16 contiguous bytes)
// B (32x16 f16) from an N-major LDS tile (row n, contiguous k):
//   lane n=lane&15 is column N=n; elements 0..7 hold K = 8h..8h+7, 8..15 hold K = 16+8h..16+8h+7
// C/D (16x16 f32): VGPR r: row M = 8h + r, col N = lane&15.
// ---------------------------------------------------------------------------
__device__ __forceinline__ v16h frag_a16(const _Float16* t, int row0, int ld, int k0) {
  const int lane = threadIdx.x & 31;
  const int r = lane & 15, h = lane >> 4;
  const _Float16* p = t + (row0 + r) * ld + k0;
  F16x16 u;
  u.f4[0] = *(const float4*)(p + 8 * h);
  u.f4[1] = *(const float4*)(p + 16 + 8 * h);
  return u.v;
}

__device__ __forceinline__ v16h frag_b16(const _Float16* t, int row0, int ld, int k0) {
  const int lane = threadIdx.x & 31;
  const int n = lane & 15, h = lane >> 4;
  const _Float16* p = t + (row0 + n) * ld + k0;
  F16x16 u;
  u.f4[0] = *(const float4*)(p + 8 * h);
  u.f4[1] = *(const float4*)(p + 16 + 8 * h);
  return u.v;
}

// ---------------------------------------------------------------------------
// LayerNorm: one 256-thread block per row of E=1024, writes f16.
// ---------------------------------------------------------------------------
__global__ __launch_bounds__(256) void ln_kernel(const float* __restrict__ x,
                                                 const float* __restrict__ g,
                                                 const float* __restrict__ b,
                                                 _Float16* __restrict__ out) {
  const int row = blockIdx.x;
  const float* xr = x + (size_t)row * 1024;
  const int t = threadIdx.x;
  float v[4];
  float s = 0.f;
#pragma unroll
  for (int i = 0; i < 4; ++i) { v[i] = xr[t + 256 * i]; s += v[i]; }
  __shared__ float red[8], red2[8];
#pragma unroll
  for (int m = 16; m >= 1; m >>= 1) s += __shfl_xor(s, m, 32);
  if ((t & 31) == 0) red[t >> 5] = s;
  __syncthreads();
  float tot = 0.f;
#pragma unroll
  for (int i = 0; i < 8; ++i) tot += red[i];
  const float mu = tot * (1.0f / 1024.0f);
  float s2 = 0.f;
#pragma unroll
  for (int i = 0; i < 4; ++i) { const float d = v[i] - mu; s2 += d * d; }
#pragma unroll
  for (int m = 16; m >= 1; m >>= 1) s2 += __shfl_xor(s2, m, 32);
  if ((t & 31) == 0) red2[t >> 5] = s2;
  __syncthreads();
  float tot2 = 0.f;
#pragma unroll
  for (int i = 0; i < 8; ++i) tot2 += red2[i];
  const float rs = rsqrtf(tot2 * (1.0f / 1024.0f) + 1e-5f);
  _Float16* o = out + (size_t)row * 1024;
#pragma unroll
  for (int i = 0; i < 4; ++i) {
    const int c = t + 256 * i;
    o[c] = (_Float16)((v[i] - mu) * rs * g[c] + b[c]);
  }
}

// ---------------------------------------------------------------------------
// Weight repacks (f32 -> f16, N-major / transposed so GEMM B tiles are
// contiguous row copies).
// ---------------------------------------------------------------------------
__global__ void pack_qkv_kernel(const float* __restrict__ w, _Float16* __restrict__ out) {
  const int i = blockIdx.x * blockDim.x + threadIdx.x;   // over E*E
  const int c = i >> 10, e = i & 1023;                   // out col c = h*64+d
  out[(size_t)c * 1024 + e] = (_Float16)w[((size_t)(c >> 6) * 1024 + e) * 64 + (c & 63)];
}

__global__ void pack_T_kernel(const float* __restrict__ w, _Float16* __restrict__ out,
                              int K, int N) {
  const int i = blockIdx.x * blockDim.x + threadIdx.x;   // over N*K, out[n*K+k]
  const int n = i / K, k = i - n * K;
  out[i] = (_Float16)w[(size_t)k * N + n];
}

// v [b][t][h][d] (f16) -> vt [b][h][d][t] (f16), so PV WMMA B tiles are contiguous.
__global__ void transpose_v_kernel(const _Float16* __restrict__ v, _Float16* __restrict__ vt) {
  const int i = blockIdx.x * blockDim.x + threadIdx.x;   // over 8M
  const int d = i & 63, h = (i >> 6) & 15, t = (i >> 10) & 1023, b = i >> 20;
  vt[(((size_t)(b * 16 + h) * 64 + d) << 10) | t] = v[i];
}

// ---------------------------------------------------------------------------
// Tiled WMMA GEMM: C[M,N] = A[M,K] * Bt[N,K]^T, f16 in / f32 accumulate.
// 128 threads = 4 waves; 128x64 C tile per block; wave = 32(M) x 64(N);
// K-step 32; double-buffered async LDS staging (6 b128 chunks/thread/stage).
// ---------------------------------------------------------------------------
template <bool BIAS, bool RELU, bool RES, bool F16OUT>
__global__ __launch_bounds__(128) void gemm_wmma(const _Float16* __restrict__ A,
                                                 const _Float16* __restrict__ Bt,
                                                 const float* __restrict__ bias,
                                                 const float* __restrict__ res,
                                                 float* __restrict__ out32,
                                                 _Float16* __restrict__ out16,
                                                 int M, int N, int K) {
  __shared__ _Float16 At[2][128 * 40];
  __shared__ _Float16 Bl[2][64 * 40];
  const int n0 = blockIdx.x * 64, m0 = blockIdx.y * 128;
  const int lane = threadIdx.x & 31, w = threadIdx.x >> 5;
  const int ln = lane & 15, lh = lane >> 4;
  v8f acc[2][4] = {};
  const _Float16* Ag = A + (size_t)m0 * K;
  const _Float16* Bg = Bt + (size_t)n0 * K;

  // stage one 128x32 A slice + 64x32 B slice into buffer `buf`
  auto stage = [&](int buf, int k0) {
#pragma unroll
    for (int it = 0; it < 6; ++it) {
      const int c = threadIdx.x + 128 * it;             // 0..767
      if (c < 512) {                                    // A: 128 rows x 4 chunks
        const int row = c >> 2, ko = (c & 3) * 8;
        cp16(At[buf] + row * 40 + ko, Ag + (size_t)row * K + k0 + ko);
      } else {                                          // B: 64 rows x 4 chunks
        const int cc = c - 512;
        const int row = cc >> 2, ko = (cc & 3) * 8;
        cp16(Bl[buf] + row * 40 + ko, Bg + (size_t)row * K + k0 + ko);
      }
    }
  };

  stage(0, 0);
  int pb = 0;
  for (int k0 = 0; k0 < K; k0 += 32) {
    const bool hasnext = (k0 + 32) < K;
    if (hasnext) {
      stage(pb ^ 1, k0 + 32);
      __builtin_prefetch(Ag + (size_t)(threadIdx.x) * K + k0 + 64, 0, 0);
      wait_async<6>();   // previous batch (this buffer) complete; next in flight
    } else {
      wait_async<0>();
    }
    __syncthreads();
    const v16h a0 = frag_a16(At[pb], 32 * w, 40, 0);
    const v16h a1 = frag_a16(At[pb], 32 * w + 16, 40, 0);
#pragma unroll
    for (int nt = 0; nt < 4; ++nt) {
      const v16h bfr = frag_b16(Bl[pb], nt * 16, 40, 0);
      acc[0][nt] = WMMA_F16(a0, bfr, acc[0][nt]);
      acc[1][nt] = WMMA_F16(a1, bfr, acc[1][nt]);
    }
    __syncthreads();   // all waves done reading before buffer is re-staged
    pb ^= 1;
  }
#pragma unroll
  for (int mi = 0; mi < 2; ++mi)
#pragma unroll
    for (int nt = 0; nt < 4; ++nt) {
      const int n = n0 + nt * 16 + ln;
      const float bv = BIAS ? bias[n] : 0.f;
#pragma unroll
      for (int r = 0; r < 8; ++r) {
        const int m = m0 + 32 * w + 16 * mi + 8 * lh + r;
        float v = acc[mi][nt][r] + bv;
        if (RELU) v = fmaxf(v, 0.f);
        if (RES) v += res[(size_t)m * N + n];
        if (F16OUT) out16[(size_t)m * N + n] = (_Float16)v;
        else        out32[(size_t)m * N + n] = v;
      }
    }
}

// ---------------------------------------------------------------------------
// Flash-style causal attention per (b, h, 64-row i block).
// Reference: scores = K·Qᵀ / E², mask j>i, softmax over j, O = P·V.
// ---------------------------------------------------------------------------
__device__ __forceinline__ void stage_tile64(const _Float16* __restrict__ g, int ldg,
                                             _Float16* __restrict__ s) {
#pragma unroll
  for (int it = 0; it < 4; ++it) {
    const int c = threadIdx.x + 128 * it;   // 512 chunks
    const int row = c >> 3, ko = (c & 7) * 8;
    cp16(s + row * 72 + ko, g + (size_t)row * ldg + ko);
  }
}

__global__ __launch_bounds__(128) void attn_kernel(const _Float16* __restrict__ q,
                                                   const _Float16* __restrict__ kk,
                                                   const _Float16* __restrict__ vt,
                                                   _Float16* __restrict__ o) {
  __shared__ _Float16 Kt[64 * 72];
  __shared__ _Float16 Qt[64 * 72];
  __shared__ _Float16 Vt[64 * 72];
  __shared__ _Float16 Pt[64 * 72];
  const int ib = blockIdx.x;
  const int bh = blockIdx.y, b = bh >> 4, h = bh & 15;
  const int i0 = ib * 64;
  const int lane = threadIdx.x & 31, w = threadIdx.x >> 5;
  const int ln = lane & 15, lh = lane >> 4;

  stage_tile64(kk + ((size_t)(b * 1024 + i0) * 1024) + h * 64, 1024, Kt);

  v8f accO[4] = {};
  float mrun[8], lrun[8];
#pragma unroll
  for (int r = 0; r < 8; ++r) { mrun[r] = -__builtin_inff(); lrun[r] = 0.f; }
  const float scale = 1.0f / (1024.0f * 1024.0f);

  for (int jb = 0; jb <= ib; ++jb) {
    const int j0 = jb * 64;
    __syncthreads();
    stage_tile64(q + ((size_t)(b * 1024 + j0) * 1024) + h * 64, 1024, Qt);
    stage_tile64(vt + ((size_t)bh * 64) * 1024 + j0, 1024, Vt);
    wait_async<0>();
    __syncthreads();

    v8f accS[4] = {};
#pragma unroll
    for (int d0 = 0; d0 < 64; d0 += 32) {
      const v16h a = frag_a16(Kt, 16 * w, 72, d0);
#pragma unroll
      for (int nt = 0; nt < 4; ++nt) {
        const v16h bf = frag_b16(Qt, nt * 16, 72, d0);
        accS[nt] = WMMA_F16(a, bf, accS[nt]);
      }
    }
    const bool diag = (jb == ib);
#pragma unroll
    for (int nt = 0; nt < 4; ++nt)
#pragma unroll
      for (int r = 0; r < 8; ++r) {
        float sv = accS[nt][r] * scale;
        if (diag) {
          const int jg = j0 + nt * 16 + ln;
          const int ig = i0 + 16 * w + 8 * lh + r;
          if (jg > ig) sv = -__builtin_inff();
        }
        accS[nt][r] = sv;
      }
    // online softmax; row (8*lh + r) lives across the 16 lanes of one half
#pragma unroll
    for (int r = 0; r < 8; ++r) {
      float mx = fmaxf(fmaxf(accS[0][r], accS[1][r]), fmaxf(accS[2][r], accS[3][r]));
      mx = fmaxf(mx, __shfl_xor(mx, 1, 32));
      mx = fmaxf(mx, __shfl_xor(mx, 2, 32));
      mx = fmaxf(mx, __shfl_xor(mx, 4, 32));
      mx = fmaxf(mx, __shfl_xor(mx, 8, 32));
      const float mnew = fmaxf(mrun[r], mx);
      float s = 0.f;
#pragma unroll
      for (int nt = 0; nt < 4; ++nt) {
        const float p = __expf(accS[nt][r] - mnew);
        accS[nt][r] = p;
        s += p;
      }
      s += __shfl_xor(s, 1, 32);
      s += __shfl_xor(s, 2, 32);
      s += __shfl_xor(s, 4, 32);
      s += __shfl_xor(s, 8, 32);
      const float alpha = __expf(mrun[r] - mnew);
      lrun[r] = lrun[r] * alpha + s;
      mrun[r] = mnew;
#pragma unroll
      for (int nt = 0; nt < 4; ++nt) accO[nt][r] *= alpha;
    }
    // spill P to wave-private LDS rows, re-read as A fragments (DS in-order per wave)
#pragma unroll
    for (int nt = 0; nt < 4; ++nt)
#pragma unroll
      for (int r = 0; r < 8; ++r)
        Pt[(16 * w + 8 * lh + r) * 72 + nt * 16 + ln] = (_Float16)accS[nt][r];
#pragma unroll
    for (int kj = 0; kj < 64; kj += 32) {
      const v16h pa = frag_a16(Pt, 16 * w, 72, kj);
#pragma unroll
      for (int nt = 0; nt < 4; ++nt) {
        const v16h vb = frag_b16(Vt, nt * 16, 72, kj);
        accO[nt] = WMMA_F16(pa, vb, accO[nt]);
      }
    }
  }
  _Float16* ob = o + ((size_t)(b * 1024 + i0 + 16 * w) * 1024) + h * 64;
#pragma unroll
  for (int nt = 0; nt < 4; ++nt)
#pragma unroll
    for (int r = 0; r < 8; ++r) {
      const float v = accO[nt][r] / lrun[r];
      ob[(size_t)(8 * lh + r) * 1024 + nt * 16 + ln] = (_Float16)v;
    }
}

// ---------------------------------------------------------------------------
// Host side
// ---------------------------------------------------------------------------
extern "C" void kernel_launch(void* const* d_in, const int* in_sizes, int n_in,
                              void* d_out, int out_size, void* d_ws, size_t ws_size,
                              hipStream_t stream) {
  (void)in_sizes; (void)n_in; (void)out_size; (void)ws_size;
  const float* x      = (const float*)d_in[0];
  const float* wq     = (const float*)d_in[1];
  const float* wk     = (const float*)d_in[2];
  const float* wv     = (const float*)d_in[3];
  const float* w_proj = (const float*)d_in[4];
  const float* b_proj = (const float*)d_in[5];
  const float* w1     = (const float*)d_in[6];
  const float* b1     = (const float*)d_in[7];
  const float* w2     = (const float*)d_in[8];
  const float* b2     = (const float*)d_in[9];
  const float* ln1_g  = (const float*)d_in[10];
  const float* ln1_b  = (const float*)d_in[11];
  const float* ln2_g  = (const float*)d_in[12];
  const float* ln2_b  = (const float*)d_in[13];
  float* out = (float*)d_out;

  const int M = 8192, E = 1024, F = 4096;
  const size_t ME = (size_t)M * E;          // 8M elements

  char* ws = (char*)d_ws;
  _Float16* qb   = (_Float16*)(ws);                       // [0,16MB)   -- act reuses [0,64MB)
  _Float16* kb   = (_Float16*)(ws + 16ull * 1024 * 1024);
  _Float16* vb   = (_Float16*)(ws + 32ull * 1024 * 1024);
  _Float16* vtw  = (_Float16*)(ws + 48ull * 1024 * 1024);
  _Float16* act  = (_Float16*)(ws);                        // M*F f16 = 64MB (reuse)
  _Float16* h1   = (_Float16*)(ws + 64ull * 1024 * 1024);  // also reused as h2
  _Float16* attn = (_Float16*)(ws + 80ull * 1024 * 1024);
  float*    x2   = (float*)   (ws + 96ull * 1024 * 1024);  // 32MB f32
  _Float16* wqp  = (_Float16*)(ws + 128ull * 1024 * 1024);
  _Float16* wkp  = (_Float16*)(ws + 130ull * 1024 * 1024);
  _Float16* wvp  = (_Float16*)(ws + 132ull * 1024 * 1024);
  _Float16* wpT  = (_Float16*)(ws + 134ull * 1024 * 1024);
  _Float16* w1T  = (_Float16*)(ws + 136ull * 1024 * 1024); // 8MB
  _Float16* w2T  = (_Float16*)(ws + 144ull * 1024 * 1024); // 8MB

  // 1. LN1
  ln_kernel<<<M, 256, 0, stream>>>(x, ln1_g, ln1_b, h1);
  // 2. weight repacks (f16, N-major)
  pack_qkv_kernel<<<(E * E) / 256, 256, 0, stream>>>(wq, wqp);
  pack_qkv_kernel<<<(E * E) / 256, 256, 0, stream>>>(wk, wkp);
  pack_qkv_kernel<<<(E * E) / 256, 256, 0, stream>>>(wv, wvp);
  pack_T_kernel<<<(E * E) / 256, 256, 0, stream>>>(w_proj, wpT, E, E);
  pack_T_kernel<<<(E * F) / 256, 256, 0, stream>>>(w1, w1T, E, F);
  pack_T_kernel<<<(F * E) / 256, 256, 0, stream>>>(w2, w2T, F, E);
  // 3. QKV projections
  dim3 gE(E / 64, M / 128), blk(128);
  gemm_wmma<false, false, false, true><<<gE, blk, 0, stream>>>(
      h1, wqp, nullptr, nullptr, nullptr, qb, M, E, E);
  gemm_wmma<false, false, false, true><<<gE, blk, 0, stream>>>(
      h1, wkp, nullptr, nullptr, nullptr, kb, M, E, E);
  gemm_wmma<false, false, false, true><<<gE, blk, 0, stream>>>(
      h1, wvp, nullptr, nullptr, nullptr, vb, M, E, E);
  // 4. V -> [B,H,HD,T]
  transpose_v_kernel<<<(int)(ME / 256), 256, 0, stream>>>(vb, vtw);
  // 5. attention
  attn_kernel<<<dim3(16, 128), blk, 0, stream>>>(qb, kb, vtw, attn);
  // 6. out projection + residual -> x2 (f32)
  gemm_wmma<true, false, true, false><<<gE, blk, 0, stream>>>(
      attn, wpT, b_proj, x, x2, nullptr, M, E, E);
  // 7. LN2 (h2 reuses h1 buffer)
  ln_kernel<<<M, 256, 0, stream>>>(x2, ln2_g, ln2_b, h1);
  // 8. FFN1 (+bias, relu) -> act f16
  gemm_wmma<true, true, false, true><<<dim3(F / 64, M / 128), blk, 0, stream>>>(
      h1, w1T, b1, nullptr, nullptr, act, M, F, E);
  // 9. FFN2 (+bias, +x2 residual) -> f32 out
  gemm_wmma<true, false, true, false><<<gE, blk, 0, stream>>>(
      act, w2T, b2, x2, out, nullptr, M, E, F);
}
